// CausalSelfAttention_60610578481678
// MI455X (gfx1250) — matrix operand order repaired and checked by reading, not compiled, not run on hardware
//
#include <hip/hip_runtime.h>

#define SEQ 4096
#define DIN 1024
#define DOUT 1024

typedef __attribute__((ext_vector_type(16))) __bf16 v16bf;
typedef __attribute__((ext_vector_type(8)))  float  v8f;

union FragBF16 {
    uint4 u[2];
    v16bf v;
};

__device__ __forceinline__ unsigned short f2bf(float x) {
    unsigned u = __builtin_bit_cast(unsigned, x);
    u += 0x7FFFu + ((u >> 16) & 1u);           // round-to-nearest-even
    return (unsigned short)(u >> 16);
}
__device__ __forceinline__ float bf2f(unsigned h) {
    unsigned u = (h & 0xFFFFu) << 16;
    return __builtin_bit_cast(float, u);
}

// ---- CDNA5 async global->LDS (ASYNCcnt path) via inline asm (portable
// across ROCm-7.2 / amdgpu-toolchain builtin signature differences).
// vdst = 32-bit LDS byte offset, vaddr = 64-bit global address.
__device__ __forceinline__ void async_g2l_b128(void* lds_dst, const void* gsrc) {
    unsigned loff = (unsigned)(unsigned long long)(uintptr_t)lds_dst; // LDS offset = low 32 bits
    unsigned long long gaddr = (unsigned long long)(uintptr_t)gsrc;
    asm volatile("global_load_async_to_lds_b128 %0, %1, off"
                 :: "v"(loff), "v"(gaddr)
                 : "memory");
}
__device__ __forceinline__ void wait_asynccnt0() {
#if __has_builtin(__builtin_amdgcn_s_wait_asynccnt)
    __builtin_amdgcn_s_wait_asynccnt(0);
#else
    asm volatile("s_wait_asynccnt 0" ::: "memory");
#endif
}

// ---------------------------------------------------------------- converts
__global__ void cvt_f32_bf16(const float* __restrict__ in,
                             unsigned short* __restrict__ out, int n) {
    int i = blockIdx.x * blockDim.x + threadIdx.x;
    if (i < n) out[i] = f2bf(in[i]);
}

// W[k][n] f32 (1024x1024) -> Wt[n][k] bf16
__global__ void cvt_transpose_bf16(const float* __restrict__ w,
                                   unsigned short* __restrict__ wt) {
    int i = blockIdx.x * blockDim.x + threadIdx.x;   // over 1M elements
    int n = i >> 10;
    int k = i & 1023;
    wt[(size_t)n * 1024 + k] = f2bf(w[(size_t)k * 1024 + n]);
}

// ---------------------------------------------------------------- WMMA GEMM
// C(MxN) = A(MxK bf16 row-major) * B(KxN), B supplied as Bt(NxK bf16).
// Block tile 64x256: 8 waves in 2x4, each wave 32x64 (2x4 wmma accums).
// K staged through LDS in BK=32 slices, double buffered, async loads.
enum { EPI_BF16_SCALED = 0, EPI_BF16_TRANS = 1, EPI_EXP_CAUSAL = 2, EPI_F32_DIVROW = 3 };

#define BK   32
#define PADU 8                  // pad ushorts per row: row stride 40 us = 80 B
#define AROW (BK + PADU)        // 40
#define BROW (BK + PADU)        // 40

template <int EPI>
__global__ __launch_bounds__(256)
void gemm_wmma(const unsigned short* __restrict__ A,
               const unsigned short* __restrict__ Bt,
               void* __restrict__ Out,
               const float* __restrict__ rrow,
               int M, int N, int K, float scale) {
    __shared__ unsigned short ldsA[2][64  * AROW];   // 2 x  5.0 KB
    __shared__ unsigned short ldsB[2][256 * BROW];   // 2 x 20.0 KB

    const int tid  = threadIdx.x;
    const int wave = tid >> 5;
    const int lane = tid & 31;
    const int lr   = lane & 15;
    const int lh   = lane >> 4;

    const int bm0 = blockIdx.y * 64;
    const int bn0 = blockIdx.x * 256;
    const int m0  = bm0 + (wave >> 2) * 32;
    const int n0  = bn0 + (wave & 3) * 64;

    v8f acc[2][4] = {};

    bool skip = false;
    if (EPI == EPI_EXP_CAUSAL) {
        skip = (bn0 > bm0 + 63);            // block fully above causal diagonal
    }

    if (!skip) {
        // per-thread async-copy chunk coordinates (16 B chunks, 4 per row)
        const int arow = tid >> 2, acol = tid & 3;                 // 1 A chunk
        const int brow0 = tid >> 2, bcol = tid & 3;                // 4 B chunks

        const unsigned short* gA = A  + (size_t)(bm0 + arow) * K + acol * 8;
        const unsigned short* gB = Bt + (size_t)(bn0 + brow0) * K + bcol * 8;

        // prefetch stage 0
        {
            async_g2l_b128(&ldsA[0][arow * AROW + acol * 8], gA);
#pragma unroll
            for (int i = 0; i < 4; ++i)
                async_g2l_b128(&ldsB[0][(brow0 + i * 64) * BROW + bcol * 8],
                               gB + (size_t)i * 64 * K);
        }
        wait_asynccnt0();
        __syncthreads();

        const int nkt = K / BK;
        for (int kt = 0; kt < nkt; ++kt) {
            const int cur = kt & 1;
            if (kt + 1 < nkt) {
                const int nxt = cur ^ 1;
                const size_t ko = (size_t)(kt + 1) * BK;
                async_g2l_b128(&ldsA[nxt][arow * AROW + acol * 8], gA + ko);
#pragma unroll
                for (int i = 0; i < 4; ++i)
                    async_g2l_b128(&ldsB[nxt][(brow0 + i * 64) * BROW + bcol * 8],
                                   gB + (size_t)i * 64 * K + ko);
            }

            // fragment reads from LDS (ds_load_b128, conflict-free padding)
            FragBF16 a[2], b[4];
#pragma unroll
            for (int i = 0; i < 2; ++i) {
                const unsigned short* ap =
                    &ldsA[cur][((wave >> 2) * 32 + i * 16 + lr) * AROW];
                a[i].u[0] = *(const uint4*)(ap + 8 * lh);
                a[i].u[1] = *(const uint4*)(ap + 16 + 8 * lh);
            }
#pragma unroll
            for (int j = 0; j < 4; ++j) {
                const unsigned short* bp =
                    &ldsB[cur][((wave & 3) * 64 + j * 16 + lr) * BROW];
                b[j].u[0] = *(const uint4*)(bp + 8 * lh);
                b[j].u[1] = *(const uint4*)(bp + 16 + 8 * lh);
            }
#pragma unroll
            for (int i = 0; i < 2; ++i)
#pragma unroll
                for (int j = 0; j < 4; ++j)
                    acc[i][j] = __builtin_amdgcn_wmma_f32_16x16x32_bf16(
                        false, a[i].v, false, b[j].v,
                        (short)0, acc[i][j], false, false);

            wait_asynccnt0();       // next stage resident
            __syncthreads();        // all waves done with 'cur' before refill
        }
    }

    // Epilogue. C/D layout: vgpr e, lane L -> row = e + 8*(L/16), col = L%16.
#pragma unroll
    for (int i = 0; i < 2; ++i) {
#pragma unroll
        for (int j = 0; j < 4; ++j) {
#pragma unroll
            for (int e = 0; e < 8; ++e) {
                const int row = m0 + i * 16 + e + 8 * lh;
                const int col = n0 + j * 16 + lr;
                const float v = acc[i][j][e];
                if (EPI == EPI_BF16_SCALED) {
                    ((unsigned short*)Out)[(size_t)row * N + col] = f2bf(v * scale);
                } else if (EPI == EPI_BF16_TRANS) {
                    ((unsigned short*)Out)[(size_t)col * M + row] = f2bf(v);
                } else if (EPI == EPI_EXP_CAUSAL) {
                    const float p = (col <= row) ? __expf(v) : 0.0f;
                    ((unsigned short*)Out)[(size_t)row * N + col] = f2bf(p);
                } else { // EPI_F32_DIVROW
                    ((float*)Out)[(size_t)row * N + col] = v * rrow[row];
                }
            }
        }
    }
}

// ---------------------------------------------------------------- row sums
__global__ __launch_bounds__(256)
void rowsum_recip(const unsigned short* __restrict__ P, float* __restrict__ rs) {
    const int row  = blockIdx.x * 8 + (threadIdx.x >> 5);
    const int lane = threadIdx.x & 31;
    const uint4* p4 = (const uint4*)(P + (size_t)row * SEQ);
    float s = 0.0f;
    for (int c = lane; c < SEQ / 8; c += 32) {
        uint4 q = p4[c];
        s += bf2f(q.x) + bf2f(q.x >> 16) + bf2f(q.y) + bf2f(q.y >> 16) +
             bf2f(q.z) + bf2f(q.z >> 16) + bf2f(q.w) + bf2f(q.w >> 16);
    }
#pragma unroll
    for (int o = 16; o > 0; o >>= 1) s += __shfl_down(s, o, 32);
    if (lane == 0) rs[row] = 1.0f / s;
}

// ---------------------------------------------------------------- launch
extern "C" void kernel_launch(void* const* d_in, const int* in_sizes, int n_in,
                              void* d_out, int out_size, void* d_ws, size_t ws_size,
                              hipStream_t stream) {
    const float* x  = (const float*)d_in[0];
    const float* Wq = (const float*)d_in[1];
    const float* Wk = (const float*)d_in[2];
    const float* Wv = (const float*)d_in[3];
    float* out = (float*)d_out;

    char* ws = (char*)d_ws;
    const size_t MB = 1ull << 20;
    unsigned short* xb  = (unsigned short*)(ws + 0);        //  8 MB x bf16
    unsigned short* wqt = (unsigned short*)(ws + 8  * MB);  //  2 MB Wq^T bf16
    unsigned short* wkt = (unsigned short*)(ws + 10 * MB);  //  2 MB
    unsigned short* wvt = (unsigned short*)(ws + 12 * MB);  //  2 MB
    unsigned short* Qb  = (unsigned short*)(ws + 14 * MB);  //  8 MB Q bf16 (pre-scaled)
    unsigned short* Kb  = (unsigned short*)(ws + 22 * MB);  //  8 MB K bf16
    unsigned short* Vt  = (unsigned short*)(ws + 30 * MB);  //  8 MB V^T bf16 [D][S]
    unsigned short* Pt  = (unsigned short*)(ws + 38 * MB);  // 32 MB exp(scores) bf16
    float*          rs  = (float*)         (ws + 70 * MB);  // 16 KB row reciprocals

    dim3 blk(256);

    cvt_f32_bf16<<<(SEQ * DIN) / 256, blk, 0, stream>>>(x, xb, SEQ * DIN);
    cvt_transpose_bf16<<<(DIN * DOUT) / 256, blk, 0, stream>>>(Wq, wqt);
    cvt_transpose_bf16<<<(DIN * DOUT) / 256, blk, 0, stream>>>(Wk, wkt);
    cvt_transpose_bf16<<<(DIN * DOUT) / 256, blk, 0, stream>>>(Wv, wvt);

    dim3 gproj(DOUT / 256, SEQ / 64);
    gemm_wmma<EPI_BF16_SCALED><<<gproj, blk, 0, stream>>>(xb, wqt, Qb, nullptr,
                                                          SEQ, DOUT, DIN, 0.03125f);
    gemm_wmma<EPI_BF16_SCALED><<<gproj, blk, 0, stream>>>(xb, wkt, Kb, nullptr,
                                                          SEQ, DOUT, DIN, 1.0f);
    gemm_wmma<EPI_BF16_TRANS ><<<gproj, blk, 0, stream>>>(xb, wvt, Vt, nullptr,
                                                          SEQ, DOUT, DIN, 1.0f);

    dim3 gscore(SEQ / 256, SEQ / 64);
    gemm_wmma<EPI_EXP_CAUSAL><<<gscore, blk, 0, stream>>>(Qb, Kb, Pt, nullptr,
                                                          SEQ, SEQ, DIN, 1.0f);

    rowsum_recip<<<SEQ / 8, blk, 0, stream>>>(Pt, rs);

    gemm_wmma<EPI_F32_DIVROW><<<gproj, blk, 0, stream>>>(Pt, Vt, out, rs,
                                                         SEQ, DOUT, SEQ, 1.0f);
}
